// RahmanDynamicNet_21534966022876
// MI455X (gfx1250) — compile-verified, matrix operand order checked by reading, not run on hardware
//
#include <hip/hip_runtime.h>
#include <stdint.h>

typedef __attribute__((ext_vector_type(16))) _Float16 v16h;
typedef __attribute__((ext_vector_type(8)))  float    v8f;
typedef __attribute__((ext_vector_type(4)))  unsigned v4u;
typedef __attribute__((ext_vector_type(8)))  int      v8i;
typedef __attribute__((ext_vector_type(4)))  int      v4i;

#define Tn 4000
#define Fn 34
#define Kdim 5
#define Hn 20
#define Bn 128
#define KK 170        // Fn*Kdim, GEMM K
#define TCHUNK 128    // t-values per block = 8 m-tiles of 16 (one per wave)
#define SLABW 136     // TCHUNK + 8 halo columns (t0-2 .. t0+133)
#define SLABSZ (Fn * SLABW)   // 4624 floats, TDM tile
#define ZPAD 160      // zero pad: invalid-K reads land here (tloc <= 127)
#define VTHR 1000.0f

__device__ __forceinline__ float sigmoidf(float x) {
    return 1.0f / (1.0f + __expf(-x));
}

// ---------------------------------------------------------------------------
// Kernel 1: conv-as-GEMM via WMMA + BN + sigmoid epilogue.
//   y[t*(B*H) + b*H + h] = sigmoid(BN(conv(x)[b,h,t]))       (fp32, [T, B*H])
// Block = 256 threads (8 waves). Block handles (batch b, 128 t-values).
// x slab is DMA'd to LDS by the Tensor Data Mover; weights are packed into
// B-fragment layout (f16) in LDS while the TDM runs.
// ---------------------------------------------------------------------------
__global__ void __launch_bounds__(256)
conv_bn_sig_wmma(const float* __restrict__ x,  const float* __restrict__ cw,
                 const float* __restrict__ cb, const float* __restrict__ gam,
                 const float* __restrict__ bet,const float* __restrict__ mu,
                 const float* __restrict__ var,float* __restrict__ y)
{
    __shared__ float slab[SLABSZ + ZPAD];                     // 19,136 B fp32
    __shared__ __align__(32) _Float16 bfrag[2][6][32][16];    // 12,288 B f16

    const int tid  = threadIdx.x;
    const int lane = tid & 31;
    const int wv   = tid >> 5;
    const int b    = blockIdx.y;
    const int t0   = blockIdx.x * TCHUNK;
    const int tstart = t0 - 2;   // slab column 0 == time index tstart

    // ---- TDM: DMA x[b, 0:34, tstart : tstart+136] (fp32) into slab --------
    if (wv == 0) {
        long long gaddr = (long long)(uintptr_t)x
                        + ((long long)b * Fn * Tn + tstart) * 4;
        unsigned ldsoff = (unsigned)(uintptr_t)&slab[0];
        unsigned tdim0  = (unsigned)(Tn - tstart);   // right-edge OOB -> zeros
        v4u g0;
        g0.x = 1u;                                        // count = 1
        g0.y = ldsoff;                                    // lds_addr
        g0.z = (unsigned)(gaddr & 0xFFFFFFFFll);          // global_addr lo
        g0.w = (unsigned)((gaddr >> 32) & 0x1FFFFFFll)    // global_addr hi
             | (2u << 30);                                // type = 2 (image)
        v8i g1;
        g1[0] = (int)(2u << 16);                          // data_size = 4B
        g1[1] = (int)((tdim0 & 0xFFFFu) << 16);           // tensor_dim0[15:0]
        g1[2] = (int)(((tdim0 >> 16) & 0xFFFFu)           // tensor_dim0[31:16]
                      | ((unsigned)Fn << 16));            // tensor_dim1 = 34
        g1[3] = (int)((unsigned)SLABW << 16);             // tile_dim0 = 136
        g1[4] = Fn;                                       // tile_dim1 = 34
        g1[5] = Tn;                                       // tensor_dim0_stride
        g1[6] = 0; g1[7] = 0;
        v4i gz4 = {0, 0, 0, 0};                           // 2-D: groups 2/3 off
        v8i gz8 = {0, 0, 0, 0, 0, 0, 0, 0};
        __builtin_amdgcn_tensor_load_to_lds(g0, g1, gz4, gz4, gz8, 0);
    }

    // zero pad region (outside the TDM tile, so no ordering hazard)
    if (tid < ZPAD) slab[SLABSZ + tid] = 0.0f;

    // ---- stage B fragments (weights -> f16, pre-packed WMMA layout) -------
    // lane l holds n = l%16; element e holds k = ks*32 + (e&8)*2 + (l>>4)*8 + (e&7)
    for (int i = tid; i < 2 * 6 * 32 * 16; i += 256) {
        int nt = i / 3072;
        int r  = i - nt * 3072;
        int ks = r >> 9;
        int ln = (r >> 4) & 31;
        int e  = r & 15;
        int n  = nt * 16 + (ln & 15);
        int k  = ks * 32 + ((e & 8) << 1) + ((ln >> 4) << 3) + (e & 7);
        float v = 0.0f;
        if (n < Hn && k < KK) v = cw[n * KK + k];
        bfrag[nt][ks][ln][e] = (_Float16)v;
    }

    if (wv == 0) __builtin_amdgcn_s_wait_tensorcnt(0);
    __syncthreads();
    // left-edge halo (t < 0) -> zero (only block 0)
    if (tstart < 0 && tid < Fn * 2) {
        int f = tid >> 1, c = tid & 1;
        if (tstart + c < 0) slab[f * SLABW + c] = 0.0f;
    }
    __syncthreads();

    // ---- GEMM: each wave owns one 16-row m-tile of t-values ---------------
    v8f acc0 = {};   // n = 0..15
    v8f acc1 = {};   // n = 16..31 (only 16..19 live)
    const int  tloc = wv * 16 + (lane & 15);   // A row (local t)
    const bool hi   = (lane & 16) != 0;        // lane-half selects K offset +8
    const float* __restrict__ srow = slab + tloc;
    #pragma unroll
    for (int ks = 0; ks < 6; ++ks) {
        union { v16h v; _Float16 h[16]; } a;
        #pragma unroll
        for (int e = 0; e < 16; ++e) {
            // Both candidate K values are compile-time constants here, so the
            // (f = k/5, kc = k%5) -> LDS offset math fully constant-folds;
            // per element this is one v_cndmask + one ds_load + pack-cvt.
            const int k0 = ks * 32 + ((e & 8) << 1) + (e & 7);  // lane-half 0
            const int k1 = k0 + 8;                              // lane-half 1
            const int c0 = (k0 < KK) ? ((k0 / Kdim) * SLABW + (k0 % Kdim))
                                     : SLABSZ;                  // -> zero pad
            const int c1 = (k1 < KK) ? ((k1 / Kdim) * SLABW + (k1 % Kdim))
                                     : SLABSZ;
            a.h[e] = (_Float16)srow[hi ? c1 : c0];
        }
        v16h b0 = *(const v16h*)&bfrag[0][ks][lane][0];
        v16h b1 = *(const v16h*)&bfrag[1][ks][lane][0];
        acc0 = __builtin_amdgcn_wmma_f32_16x16x32_f16(false, a.v, false, b0,
                                                      (short)0, acc0, false, false);
        acc1 = __builtin_amdgcn_wmma_f32_16x16x32_f16(false, a.v, false, b1,
                                                      (short)0, acc1, false, false);
    }

    // ---- BN + sigmoid epilogue, scatter to y[T, B*H] ----------------------
    int n0 = lane & 15;
    float s0 = gam[n0] * __frsqrt_rn(var[n0] + 1e-5f);
    float o0 = (cb[n0] - mu[n0]) * s0 + bet[n0];
    int n1 = 16 + n0;
    bool n1ok = (n1 < Hn);
    float s1 = 0.0f, o1 = 0.0f;
    if (n1ok) {
        s1 = gam[n1] * __frsqrt_rn(var[n1] + 1e-5f);
        o1 = (cb[n1] - mu[n1]) * s1 + bet[n1];
    }
    const int rowadd = (lane >> 4) << 3;       // C layout: M = j + 8*(lane>=16)
    #pragma unroll
    for (int j = 0; j < 8; ++j) {
        int t = t0 + wv * 16 + j + rowadd;
        if (t < Tn) {
            long long base = (long long)t * (Bn * Hn) + b * Hn;
            y[base + n0] = sigmoidf(acc0[j] * s0 + o0);
            if (n1ok) y[base + n1] = sigmoidf(acc1[j] * s1 + o1);
        }
    }
}

// ---------------------------------------------------------------------------
// Kernel 2: LIF #1, in-place scan over T. 2560 independent (b,h) lanes.
// ---------------------------------------------------------------------------
__global__ void __launch_bounds__(256)
lif1_scan(float* __restrict__ yv, const float* __restrict__ w1p)
{
    int j = blockIdx.x * 256 + threadIdx.x;
    if (j >= Bn * Hn) return;
    const float sw = sigmoidf(w1p[0]);
    float v = 0.0f;
    for (int t = 0; t < Tn; ++t) {
        long long idx = (long long)t * (Bn * Hn) + j;
        __builtin_prefetch(&yv[idx + 8 * (Bn * Hn)], 0, 1);
        float xv = yv[idx];
        v = v + (xv - v) * sw;
        if (v >= VTHR) v -= VTHR;   // soft reset (surrogate fwd = heaviside)
        yv[idx] = v;
    }
}

// ---------------------------------------------------------------------------
// Kernel 3: z[t,b] = sigmoid(dot20(v1[t,b,:], lin_w) + lin_b). Fully parallel.
// ---------------------------------------------------------------------------
__global__ void __launch_bounds__(256)
linear_sig(const float* __restrict__ v1, const float* __restrict__ lw,
           const float* __restrict__ lb, float* __restrict__ z)
{
    int gid = blockIdx.x * 256 + threadIdx.x;
    if (gid >= Tn * Bn) return;
    int t = gid >> 7;
    int b = gid & 127;
    const float* p = v1 + (long long)t * (Bn * Hn) + b * Hn;
    float acc = lb[0];
    #pragma unroll
    for (int h = 0; h < Hn; ++h) acc += p[h] * lw[h];
    z[gid] = sigmoidf(acc);      // gid == t*128 + b
}

// ---------------------------------------------------------------------------
// Kernel 4: LIF #2 scan + transpose to out[B, T].
// ---------------------------------------------------------------------------
__global__ void __launch_bounds__(128)
lif2_out(const float* __restrict__ z, const float* __restrict__ w2p,
         float* __restrict__ out)
{
    int b = threadIdx.x;
    const float sw = sigmoidf(w2p[0]);
    float v = 0.0f;
    for (int t = 0; t < Tn; ++t) {
        __builtin_prefetch(&z[(t + 8) * Bn + b], 0, 1);
        float zv = z[t * Bn + b];
        v = v + (zv - v) * sw;
        if (v >= VTHR) v -= VTHR;
        out[(long long)b * Tn + t] = v;
    }
}

// ---------------------------------------------------------------------------
extern "C" void kernel_launch(void* const* d_in, const int* in_sizes, int n_in,
                              void* d_out, int out_size, void* d_ws, size_t ws_size,
                              hipStream_t stream)
{
    const float* x   = (const float*)d_in[0];
    const float* cw  = (const float*)d_in[1];
    const float* cb  = (const float*)d_in[2];
    const float* gam = (const float*)d_in[3];
    const float* bet = (const float*)d_in[4];
    const float* mu  = (const float*)d_in[5];
    const float* var = (const float*)d_in[6];
    const float* lw  = (const float*)d_in[7];
    const float* lb  = (const float*)d_in[8];
    const float* w1  = (const float*)d_in[9];
    const float* w2  = (const float*)d_in[10];
    float* out = (float*)d_out;

    // workspace: y/v1 [T, B*H] fp32 (in-place LIF1), then z [T, B] fp32
    float* y = (float*)d_ws;                                 // 40.96 MB
    float* z = y + (size_t)Tn * Bn * Hn;                     //  2.05 MB

    dim3 g1((Tn + TCHUNK - 1) / TCHUNK, Bn);                 // (32, 128)
    conv_bn_sig_wmma<<<g1, 256, 0, stream>>>(x, cw, cb, gam, bet, mu, var, y);
    lif1_scan<<<(Bn * Hn + 255) / 256, 256, 0, stream>>>(y, w1);
    linear_sig<<<(Tn * Bn + 255) / 256, 256, 0, stream>>>(y, lw, lb, z);
    lif2_out<<<1, 128, 0, stream>>>(z, w2, out);
}
